// SelfAttention_83313775608367
// MI455X (gfx1250) — matrix operand order repaired and checked by reading, not compiled, not run on hardware
//
#include <hip/hip_runtime.h>
#include <hip/hip_bf16.h>

// ---------------------------------------------------------------------------
// Self-attention for MI455X (gfx1250): bf16 WMMA (16x16x32) with f32 accum.
// GEMM operand staging uses GLOBAL_LOAD_ASYNC_TO_LDS_B128 (ASYNCcnt-tracked,
// VGPR-free) with double-buffered LDS and s_wait_asynccnt pipelining.
// Fragment loads are batched ahead of WMMA chains so DS/VMEM latency overlaps
// the matrix pipe instead of serializing on a recycled register octet.
// ---------------------------------------------------------------------------

#define BS 4
#define NS 1024
#define DIM 2048
#define NH 16
#define DH 128
#define SCALE 0.08838834764831845f   // DH^-0.5
#define MASK_BIAS (-1e9f)

typedef __attribute__((ext_vector_type(16))) __bf16 v16bf;
typedef __attribute__((ext_vector_type(8)))  __bf16 v8bf;
typedef __attribute__((ext_vector_type(8)))  float  v8f;

union FragU { v16bf v; uint4 q[2]; };
union PackU { v8bf  v; uint4 q;    };

__device__ __forceinline__ v8f wmma_bf16(v16bf a, v16bf b, v8f c) {
  // D = A(16x32) * B(32x16) + C, f32 accumulate
  return __builtin_amdgcn_wmma_f32_16x16x32_bf16(false, a, false, b,
                                                 (short)0, c, false, false);
}

__device__ __forceinline__ v8f zero8() {
  v8f c;
#pragma unroll
  for (int i = 0; i < 8; ++i) c[i] = 0.0f;
  return c;
}

// --- CDNA5 async copy: global -> LDS, 16B per lane, ASYNCcnt-tracked -------
// GVS addressing: 64-bit SGPR base + per-lane 32-bit byte offset.
// INST_OFFSET is added to BOTH the LDS and global addresses (ISA 10.7/15.18).
__device__ __forceinline__ void async_ld128_0(unsigned lds, unsigned voff,
                                              const void* base) {
  asm volatile("global_load_async_to_lds_b128 %0, %1, %2 offset:0"
               :: "v"(lds), "v"(voff), "s"(base) : "memory");
}
__device__ __forceinline__ void async_ld128_16(unsigned lds, unsigned voff,
                                               const void* base) {
  asm volatile("global_load_async_to_lds_b128 %0, %1, %2 offset:16"
               :: "v"(lds), "v"(voff), "s"(base) : "memory");
}
__device__ __forceinline__ void wait_async0() {
  asm volatile("s_wait_asynccnt 0x0" ::: "memory");
}
// Flat shared-aperture address truncated to 32 bits == LDS byte offset.
__device__ __forceinline__ unsigned lds_off(const void* p) {
  return (unsigned)(uintptr_t)p;
}

// A operand (16x32 bf16, M x K): lane L -> row m = L%16;
// holds K chunks [k0+(L/16)*8, +8) and [k0+(L/16)*8+16, +8)   (ISA 7.12.2)
__device__ __forceinline__ v16bf loadA(const __bf16* base, int stride,
                                       int lane, int k0) {
  const __bf16* p = base + (size_t)(lane & 15) * stride + k0 + ((lane >> 4) << 3);
  FragU f;
  f.q[0] = *(const uint4*)(p);
  f.q[1] = *(const uint4*)(p + 16);
  return f.v;
}

// B operand (32x16 bf16, K x N): lane L -> col n = L%16;
// holds K chunks [k0+(L/16)*8, +8) and [k0+(L/16)*8+16, +8)   (ISA 7.12.2)
__device__ __forceinline__ v16bf loadB(const __bf16* base, int stride,
                                       int lane, int k0) {
  const __bf16* p = base + (size_t)(lane & 15) * stride + k0 + ((lane >> 4) << 3);
  FragU f;
  f.q[0] = *(const uint4*)(p);
  f.q[1] = *(const uint4*)(p + 16);
  return f.v;
}

// ---------------------------------------------------------------------------
// fp32 -> bf16 conversion, 8 elems/thread
// ---------------------------------------------------------------------------
__global__ __launch_bounds__(256) void cvt_f32_bf16(const float* __restrict__ in,
                                                    __bf16* __restrict__ out,
                                                    int n) {
  int i = (blockIdx.x * 256 + threadIdx.x) * 8;
  if (i + 8 <= n) {
    float4 a = *(const float4*)(in + i);
    float4 b = *(const float4*)(in + i + 4);
    PackU p;
    p.v[0] = (__bf16)a.x; p.v[1] = (__bf16)a.y;
    p.v[2] = (__bf16)a.z; p.v[3] = (__bf16)a.w;
    p.v[4] = (__bf16)b.x; p.v[5] = (__bf16)b.y;
    p.v[6] = (__bf16)b.z; p.v[7] = (__bf16)b.w;
    *(uint4*)(out + i) = p.q;
  }
}

// ---------------------------------------------------------------------------
// NT GEMM core: 128x128 tile/WG, 8 waves each 32x64 (2x4 WMMA tiles), K=32
// per step, async-to-LDS double buffering.
// ---------------------------------------------------------------------------
#define LSTR 40   // LDS row stride (elems): 80B -> b128-aligned, conflict-free

__device__ __forceinline__ void gemm_step(const __bf16* Abuf, const __bf16* Bbuf,
                                          int mb, int nb, int lane, v8f acc[8]) {
  // Batch ALL fragment loads first (distinct regs) so ds latency overlaps WMMA
  v16bf af0 = loadA(Abuf + (size_t)mb * LSTR, LSTR, lane, 0);
  v16bf af1 = loadA(Abuf + (size_t)(mb + 16) * LSTR, LSTR, lane, 0);
  v16bf bfr[4];
#pragma unroll
  for (int nt = 0; nt < 4; ++nt)
    bfr[nt] = loadB(Bbuf + (size_t)(nb + nt * 16) * LSTR, LSTR, lane, 0);
#pragma unroll
  for (int nt = 0; nt < 4; ++nt) {
    acc[nt]     = wmma_bf16(af0, bfr[nt], acc[nt]);
    acc[4 + nt] = wmma_bf16(af1, bfr[nt], acc[4 + nt]);
  }
}

__device__ __forceinline__ void stage_async(unsigned ldsA, unsigned ldsB,
                                            unsigned voffA, unsigned voffB,
                                            const void* aB, const void* bB) {
  async_ld128_0 (ldsA, voffA, aB);
  async_ld128_16(ldsA, voffA, aB);
  async_ld128_0 (ldsB, voffB, bB);
  async_ld128_16(ldsB, voffB, bB);
}

// ---------------------------------------------------------------------------
// QKV projection: C[4096,6144] = X[4096,2048] * Win[6144,2048]^T
// Epilogue scatters Q,K -> [b,h,s,dh] and V -> [b,h,dh,s] (transposed).
// ---------------------------------------------------------------------------
__global__ __launch_bounds__(256) void gemm_qkv(const __bf16* __restrict__ xb,
                                                const __bf16* __restrict__ winb,
                                                __bf16* __restrict__ qbuf,
                                                __bf16* __restrict__ kbuf,
                                                __bf16* __restrict__ vtb) {
  __shared__ __bf16 Ab0[128 * LSTR], Ab1[128 * LSTR];
  __shared__ __bf16 Bb0[128 * LSTR], Bb1[128 * LSTR];

  const int n0   = blockIdx.x * 128;
  const int m0   = blockIdx.y * 128;
  const int tid  = threadIdx.x;
  const int lane = tid & 31;
  const int w    = tid >> 5;
  const int mb   = (w & 3) * 32;    // wave M offset in tile
  const int nb   = (w >> 2) * 64;   // wave N offset in tile

  const int lrow  = tid >> 1;          // 0..127
  const int lhalf = (tid & 1) * 16;    // 0 or 16 (elems)

  const unsigned la0 = lds_off(Ab0 + lrow * LSTR + lhalf);
  const unsigned la1 = lds_off(Ab1 + lrow * LSTR + lhalf);
  const unsigned lb0 = lds_off(Bb0 + lrow * LSTR + lhalf);
  const unsigned lb1 = lds_off(Bb1 + lrow * LSTR + lhalf);
  unsigned voffA = (unsigned)(((m0 + lrow) * DIM + lhalf) * 2);  // bytes
  unsigned voffB = (unsigned)(((n0 + lrow) * DIM + lhalf) * 2);

  v8f acc[8];
#pragma unroll
  for (int i = 0; i < 8; ++i) acc[i] = zero8();

  stage_async(la0, lb0, voffA, voffB, xb, winb);   // K-step 0 -> buf0
  voffA += 64; voffB += 64;

#pragma unroll 1
  for (int ks = 0; ks < DIM / 64; ++ks) {          // 32 double-steps
    // even step: compute buf0, prefetch buf1
    wait_async0();
    __syncthreads();
    stage_async(la1, lb1, voffA, voffB, xb, winb);
    voffA += 64; voffB += 64;
    gemm_step(Ab0, Bb0, mb, nb, lane, acc);
    // odd step: compute buf1, prefetch buf0
    wait_async0();
    __syncthreads();
    if (ks < DIM / 64 - 1) {
      stage_async(la0, lb0, voffA, voffB, xb, winb);
      voffA += 64; voffB += 64;
    }
    gemm_step(Ab1, Bb1, mb, nb, lane, acc);
  }

  // Epilogue: C-frag element r -> (row = 8*(lane>>4)+r, col = lane&15)
  const int which = n0 >> 11;          // 0=Q 1=K 2=V
  const int hh    = (n0 >> 7) & 15;    // head
  const int b     = m0 >> 10;          // batch (tile never crosses batches)
  const int hiRow = (lane >> 4) * 8;

  if (which < 2) {
    __bf16* dst = (which == 0 ? qbuf : kbuf) + (size_t)(b * NH + hh) * NS * DH;
#pragma unroll
    for (int mt = 0; mt < 2; ++mt) {
      int sBase = (m0 & (NS - 1)) + mb + mt * 16 + hiRow;
#pragma unroll
      for (int nt = 0; nt < 4; ++nt) {
        int d = nb + nt * 16 + (lane & 15);
#pragma unroll
        for (int r = 0; r < 8; ++r)
          dst[(size_t)(sBase + r) * DH + d] = (__bf16)acc[mt * 4 + nt][r];
      }
    }
  } else {
    // V transposed: [b,h,dh,s]; 8 consecutive s per lane -> packed b128 store
    __bf16* dst = vtb + (size_t)(b * NH + hh) * DH * NS;
#pragma unroll
    for (int mt = 0; mt < 2; ++mt) {
      int sBase = (m0 & (NS - 1)) + mb + mt * 16 + hiRow;
#pragma unroll
      for (int nt = 0; nt < 4; ++nt) {
        int d = nb + nt * 16 + (lane & 15);
        PackU p;
#pragma unroll
        for (int r = 0; r < 8; ++r) p.v[r] = (__bf16)acc[mt * 4 + nt][r];
        *(uint4*)(dst + (size_t)d * NS + sBase) = p.q;
      }
    }
  }
}

// ---------------------------------------------------------------------------
// Attention: one WG per (b*16+h, 128-query tile). Each of 8 waves owns 16
// query rows end-to-end (softmax needs no cross-wave sync). Full score rows
// held as bf16 in LDS (264KB of the 320KB CDNA5 WGP LDS).
// ---------------------------------------------------------------------------
#define SSTR 1032   // 128 S rows, padded: 2064B stride -> conflict-free b128

__global__ __launch_bounds__(256) void attn_kernel(const __bf16* __restrict__ qb,
                                                   const __bf16* __restrict__ kb,
                                                   const __bf16* __restrict__ vtb,
                                                   const int* __restrict__ ids,
                                                   __bf16* __restrict__ ctx) {
  __shared__ __bf16 S[128 * SSTR];
  __shared__ float  bias_s[NS];

  const int bh   = blockIdx.x;          // b*16 + h
  const int b    = bh >> 4;
  const int h    = bh & 15;
  const int q0   = blockIdx.y * 128;
  const int tid  = threadIdx.x;
  const int lane = tid & 31;
  const int w    = tid >> 5;
  const int hiRow = (lane >> 4) * 8;

  for (int i = tid; i < NS; i += 256)
    bias_s[i] = (float)ids[b * NS + i] * MASK_BIAS;
  __syncthreads();

  // Keep Q fragments (16 rows x K=128) resident in registers
  const __bf16* qbase = qb + ((size_t)bh * NS + q0 + 16 * w) * DH;
  v16bf qf[4];
#pragma unroll
  for (int ks = 0; ks < 4; ++ks) qf[ks] = loadA(qbase, DH, lane, ks * 32);

  __bf16* Srow = S + (size_t)(16 * w) * SSTR;

  float pmax[8];
#pragma unroll
  for (int r = 0; r < 8; ++r) pmax[r] = -3.0e38f;

  // ---- Pass 1: S = (Q*K^T + bias) * scale, track per-row running max ----
  const __bf16* kball = kb + (size_t)bh * NS * DH;
#pragma unroll 1
  for (int nt = 0; nt < NS / 16; ++nt) {
    const __bf16* kbase = kball + (size_t)(nt * 16) * DH;
    if (nt + 1 < NS / 16)            // global_prefetch_b8 for next K tile
      __builtin_prefetch(kbase + 16 * DH + (lane << 4), 0, 1);
    // batch all 4 K fragments, then run the WMMA chain
    v16bf kf[4];
#pragma unroll
    for (int ks = 0; ks < 4; ++ks) kf[ks] = loadB(kbase, DH, lane, ks * 32);
    v8f c = zero8();
#pragma unroll
    for (int ks = 0; ks < 4; ++ks) c = wmma_bf16(qf[ks], kf[ks], c);

    int col = nt * 16 + (lane & 15);
    float bias = bias_s[col];
#pragma unroll
    for (int r = 0; r < 8; ++r) {
      float v = (c[r] + bias) * SCALE;
      pmax[r] = fmaxf(pmax[r], v);
      Srow[(size_t)(hiRow + r) * SSTR + col] = (__bf16)v;
    }
  }
  // row-max across the 16 lanes that share a row (xor<16 stays in-half)
#pragma unroll
  for (int r = 0; r < 8; ++r) {
    float m = pmax[r];
    m = fmaxf(m, __shfl_xor(m, 1, 32));
    m = fmaxf(m, __shfl_xor(m, 2, 32));
    m = fmaxf(m, __shfl_xor(m, 4, 32));
    m = fmaxf(m, __shfl_xor(m, 8, 32));
    pmax[r] = m;
  }

  // ---- Pass 2: P = exp(S - max), accumulate row sums ----
  float rsum[8];
#pragma unroll
  for (int r = 0; r < 8; ++r) rsum[r] = 0.0f;
#pragma unroll 1
  for (int nt = 0; nt < NS / 16; ++nt) {
    int col = nt * 16 + (lane & 15);
#pragma unroll
    for (int r = 0; r < 8; ++r) {
      __bf16* p = Srow + (size_t)(hiRow + r) * SSTR + col;
      float e = __expf((float)(*p) - pmax[r]);
      rsum[r] += e;
      *p = (__bf16)e;
    }
  }
#pragma unroll
  for (int r = 0; r < 8; ++r) {
    float s = rsum[r];
    s += __shfl_xor(s, 1, 32);
    s += __shfl_xor(s, 2, 32);
    s += __shfl_xor(s, 4, 32);
    s += __shfl_xor(s, 8, 32);
    rsum[r] = 1.0f / s;
  }

  // ---- Pass 3: O = P * V  (V stored [dh][keys] -> contiguous B frags) ----
  v8f acc[8];
#pragma unroll
  for (int i = 0; i < 8; ++i) acc[i] = zero8();
  const __bf16* vbase = vtb + (size_t)bh * DH * NS;

#pragma unroll 1
  for (int kt = 0; kt < NS / 32; ++kt) {
    v16bf af = loadA(Srow, SSTR, lane, kt * 32);
    if (kt + 1 < NS / 32)            // prefetch next V K-slab
      __builtin_prefetch(vbase + (size_t)(lane << 2) * NS + (kt + 1) * 32, 0, 1);
    // two groups of 4: batch V-fragment loads ahead of each WMMA chain
#pragma unroll
    for (int g = 0; g < 2; ++g) {
      v16bf vf[4];
#pragma unroll
      for (int j = 0; j < 4; ++j)
        vf[j] = loadB(vbase + (size_t)((g * 4 + j) * 16) * NS, NS, lane, kt * 32);
#pragma unroll
      for (int j = 0; j < 4; ++j)
        acc[g * 4 + j] = wmma_bf16(af, vf[j], acc[g * 4 + j]);
    }
  }

  // ---- Store context as [b, s, h*128+d] bf16 ----
  __bf16* cbase = ctx + ((size_t)b * NS + q0 + 16 * w) * DIM + h * DH;
#pragma unroll
  for (int nt = 0; nt < 8; ++nt) {
    int d = nt * 16 + (lane & 15);
#pragma unroll
    for (int r = 0; r < 8; ++r) {
      float v = acc[nt][r] * rsum[r];
      cbase[(size_t)(hiRow + r) * DIM + d] = (__bf16)v;
    }
  }
}

// ---------------------------------------------------------------------------
// Output projection: out[4096,2048] = ctx * Wout^T + b_out  (f32 output)
// ---------------------------------------------------------------------------
__global__ __launch_bounds__(256) void gemm_out(const __bf16* __restrict__ ctx,
                                                const __bf16* __restrict__ wob,
                                                const float* __restrict__ bias,
                                                float* __restrict__ out) {
  __shared__ __bf16 Ab0[128 * LSTR], Ab1[128 * LSTR];
  __shared__ __bf16 Bb0[128 * LSTR], Bb1[128 * LSTR];

  const int n0   = blockIdx.x * 128;
  const int m0   = blockIdx.y * 128;
  const int tid  = threadIdx.x;
  const int lane = tid & 31;
  const int w    = tid >> 5;
  const int mb   = (w & 3) * 32;
  const int nb   = (w >> 2) * 64;

  const int lrow  = tid >> 1;
  const int lhalf = (tid & 1) * 16;

  const unsigned la0 = lds_off(Ab0 + lrow * LSTR + lhalf);
  const unsigned la1 = lds_off(Ab1 + lrow * LSTR + lhalf);
  const unsigned lb0 = lds_off(Bb0 + lrow * LSTR + lhalf);
  const unsigned lb1 = lds_off(Bb1 + lrow * LSTR + lhalf);
  unsigned voffA = (unsigned)(((m0 + lrow) * DIM + lhalf) * 2);
  unsigned voffB = (unsigned)(((n0 + lrow) * DIM + lhalf) * 2);

  v8f acc[8];
#pragma unroll
  for (int i = 0; i < 8; ++i) acc[i] = zero8();

  stage_async(la0, lb0, voffA, voffB, ctx, wob);
  voffA += 64; voffB += 64;

#pragma unroll 1
  for (int ks = 0; ks < DIM / 64; ++ks) {
    wait_async0();
    __syncthreads();
    stage_async(la1, lb1, voffA, voffB, ctx, wob);
    voffA += 64; voffB += 64;
    gemm_step(Ab0, Bb0, mb, nb, lane, acc);

    wait_async0();
    __syncthreads();
    if (ks < DIM / 64 - 1) {
      stage_async(la0, lb0, voffA, voffB, ctx, wob);
      voffA += 64; voffB += 64;
    }
    gemm_step(Ab1, Bb1, mb, nb, lane, acc);
  }

  const int hiRow = (lane >> 4) * 8;
#pragma unroll
  for (int nt = 0; nt < 4; ++nt) {
    int n = n0 + nb + nt * 16 + (lane & 15);
    float bv = bias[n];
#pragma unroll
    for (int mt = 0; mt < 2; ++mt) {
      int m = m0 + mb + mt * 16 + hiRow;
#pragma unroll
      for (int r = 0; r < 8; ++r)
        out[(size_t)(m + r) * DIM + n] = acc[mt * 4 + nt][r] + bv;
    }
  }
}

// ---------------------------------------------------------------------------
// Host launcher
// ---------------------------------------------------------------------------
extern "C" void kernel_launch(void* const* d_in, const int* in_sizes, int n_in,
                              void* d_out, int out_size, void* d_ws, size_t ws_size,
                              hipStream_t stream) {
  const float* x     = (const float*)d_in[0];   // [4,1024,2048]
  const int*   ids   = (const int*)d_in[1];     // [4,1024]
  const float* w_in  = (const float*)d_in[2];   // [6144,2048]
  const float* w_out = (const float*)d_in[3];   // [2048,2048]
  const float* b_out = (const float*)d_in[4];   // [2048]
  float* out = (float*)d_out;                   // [4,1024,2048] f32

  const size_t nX  = (size_t)BS * NS * DIM;       // 8,388,608
  const size_t nWi = (size_t)3 * DIM * DIM;       // 12,582,912
  const size_t nWo = (size_t)DIM * DIM;           // 4,194,304

  __bf16* xb   = (__bf16*)d_ws;
  __bf16* winb = xb + nX;
  __bf16* wob  = winb + nWi;
  __bf16* qbuf = wob + nWo;        // [b,h,s,dh]
  __bf16* kbuf = qbuf + nX;        // [b,h,s,dh]
  __bf16* vtb  = kbuf + nX;        // [b,h,dh,s]
  __bf16* ctx  = vtb + nX;         // [b,s,h*dh]   (~112 MB total)

  cvt_f32_bf16<<<(int)(nX / 2048),  256, 0, stream>>>(x, xb, (int)nX);
  cvt_f32_bf16<<<(int)(nWi / 2048), 256, 0, stream>>>(w_in, winb, (int)nWi);
  cvt_f32_bf16<<<(int)(nWo / 2048), 256, 0, stream>>>(w_out, wob, (int)nWo);

  gemm_qkv<<<dim3(3 * DIM / 128, BS * NS / 128), 256, 0, stream>>>(
      xb, winb, qbuf, kbuf, vtb);

  attn_kernel<<<dim3(BS * NH, NS / 128), 256, 0, stream>>>(
      qbuf, kbuf, vtb, ids, ctx);

  gemm_out<<<dim3(DIM / 128, BS * NS / 128), 256, 0, stream>>>(
      ctx, wob, b_out, out);
}
